// LongformerAttention_11982958756577
// MI455X (gfx1250) — matrix operand order repaired and checked by reading, not compiled, not run on hardware
//
#include <hip/hip_runtime.h>
#include <stdint.h>

#define S_LEN 4096
#define B_SZ  2
#define E_SZ  768
#define H_SZ  12
#define D_SZ  64
#define NC    16
#define WIN   768   // 3*W window of keys per chunk
#define WHALF 256

typedef __attribute__((ext_vector_type(16))) __bf16 v16bf;
typedef __attribute__((ext_vector_type(8)))  float  v8f;

union Frag { uint32_t u[8]; v16bf v; };

__device__ __forceinline__ uint16_t f2bf(float f) {
  uint32_t u = __builtin_bit_cast(uint32_t, f);
  u += 0x7fffu + ((u >> 16) & 1u);   // round-to-nearest-even
  return (uint16_t)(u >> 16);
}
__device__ __forceinline__ uint32_t pack2(uint16_t lo, uint16_t hi) {
  return (uint32_t)lo | ((uint32_t)hi << 16);
}
__device__ __forceinline__ v8f vzero8() {
  v8f z;
#pragma unroll
  for (int i = 0; i < 8; i++) z[i] = 0.0f;
  return z;
}

// Async DMA: copy 16 bytes/lane from global (base + byte offset) into LDS at
// per-lane LDS byte address.  Tracked by ASYNCcnt.
__device__ __forceinline__ void async_ld16(uint32_t lds_byte_addr,
                                           uint32_t gl_byte_off,
                                           const void* sbase) {
  asm volatile("global_load_async_to_lds_b128 %0, %1, %2"
               :: "v"(lds_byte_addr), "v"(gl_byte_off), "s"(sbase)
               : "memory");
}
#define WAIT_ASYNC(n) asm volatile("s_wait_asynccnt %0" :: "i"(n) : "memory")

// LDS byte address of a __shared__ object = low 32 bits of its flat address.
template <typename T>
__device__ __forceinline__ uint32_t lds_addr_of(T* p) {
  return (uint32_t)(uintptr_t)p;
}

// ---------------------------------------------------------------------------
// Kernel 0a: x (S,B,E) f32 -> xb (B*S, E) bf16, GEMM row order r = b*S + s
// ---------------------------------------------------------------------------
__global__ __launch_bounds__(256) void cvt_x_kernel(
    const float* __restrict__ x, uint32_t* __restrict__ xb32)
{
  int idx = blockIdx.x * 256 + threadIdx.x;      // 3,145,728 u32 total
  int r = idx / (E_SZ / 2);
  int p = idx - r * (E_SZ / 2);
  int b = r >> 12, s = r & (S_LEN - 1);
  const float* src = x + ((size_t)s * B_SZ + b) * E_SZ + p * 2;
  __builtin_prefetch(src + 256, 0, 1);
  float2 v = *(const float2*)src;
  xb32[idx] = pack2(f2bf(v.x), f2bf(v.y));
}

// ---------------------------------------------------------------------------
// Kernel 0b: Wq/Wk/Wv (E,E) f32 -> wtb bf16 TRANSPOSED: wt[z][n][k] = W[k][n]
// ---------------------------------------------------------------------------
__global__ __launch_bounds__(256) void cvt_w_kernel(
    const float* __restrict__ Wq, const float* __restrict__ Wk,
    const float* __restrict__ Wv, uint32_t* __restrict__ wtb32)
{
  const float* wp = (blockIdx.y == 0) ? Wq : (blockIdx.y == 1) ? Wk : Wv;
  int idx = blockIdx.x * 256 + threadIdx.x;      // 294,912 u32 per matrix
  int n = idx / (E_SZ / 2);
  int kp = idx - n * (E_SZ / 2);
  float w0 = wp[(size_t)(2 * kp) * E_SZ + n];
  float w1 = wp[(size_t)(2 * kp + 1) * E_SZ + n];
  wtb32[(size_t)blockIdx.y * (E_SZ * E_SZ / 2) + idx] = pack2(f2bf(w0), f2bf(w1));
}

// ---------------------------------------------------------------------------
// Kernel 1: QKV GEMM, all-bf16 operands, async double-buffered LDS staging.
// M=8192, N=768, K=768; tile 128x128, BK=32; 8 waves of 64x32.
// z==0 -> Q (scaled), z==1 -> K, z==2 -> V stored TRANSPOSED (B,H,D,S).
// ---------------------------------------------------------------------------
__global__ __launch_bounds__(256) void qkv_gemm_kernel(
    const uint16_t* __restrict__ xb, const uint16_t* __restrict__ wtb,
    const float* __restrict__ bq, const float* __restrict__ bk,
    const float* __restrict__ bv,
    uint16_t* __restrict__ qb, uint16_t* __restrict__ kb,
    uint16_t* __restrict__ vtw)
{
  __shared__ __align__(16) uint32_t a_lds[2][128 * 16];  // [buf][row][kpair]
  __shared__ __align__(16) uint32_t b_lds[2][128 * 16];  // [buf][n][kpair]

  const float* bp; uint16_t* outp; float scale;
  if (blockIdx.z == 0)      { bp = bq; outp = qb;  scale = 0.125f; }
  else if (blockIdx.z == 1) { bp = bk; outp = kb;  scale = 1.0f;   }
  else                      { bp = bv; outp = vtw; scale = 1.0f;   }
  const uint16_t* wt = wtb + (size_t)blockIdx.z * (E_SZ * E_SZ);

  const int tid  = threadIdx.x;
  const int wave = tid >> 5;
  const int lane = tid & 31;
  const int half = lane >> 4;
  const int ml   = lane & 15;
  const int wm   = wave >> 2;   // 0..1 -> 64 rows
  const int wn   = wave & 3;    // 0..3 -> 32 cols

  const int row0 = blockIdx.x * 128;
  const int col0 = blockIdx.y * 128;

  const uint32_t aL = lds_addr_of(&a_lds[0][0]) + tid * 16;
  const uint32_t bL = lds_addr_of(&b_lds[0][0]) + tid * 16;
  const uint32_t arow = (uint32_t)(row0 + (tid >> 2)) * (E_SZ * 2) + (tid & 3) * 16;
  const uint32_t brow = (uint32_t)(col0 + (tid >> 2)) * (E_SZ * 2) + (tid & 3) * 16;

  v8f acc[4][2];
#pragma unroll
  for (int i = 0; i < 4; i++)
#pragma unroll
    for (int j = 0; j < 2; j++) acc[i][j] = vzero8();

  // prologue: stage tile 0 into buffer 0
  async_ld16(aL, arow, xb);
  async_ld16(bL, brow, wt);

  for (int j = 0; j < 24; j++) {           // 24 K-steps of 32
    const int buf = j & 1;
    if (j + 1 < 24) {
      async_ld16(aL + (buf ^ 1) * 8192, arow + (uint32_t)(j + 1) * 64, xb);
      async_ld16(bL + (buf ^ 1) * 8192, brow + (uint32_t)(j + 1) * 64, wt);
      WAIT_ASYNC(2);
    } else {
      WAIT_ASYNC(0);
    }
    __syncthreads();

    const uint32_t* ab = a_lds[buf];
    const uint32_t* bb = b_lds[buf];
    Frag afr[4], bfr[2];
#pragma unroll
    for (int mt = 0; mt < 4; mt++) {
      int row = wm * 64 + mt * 16 + ml;
#pragma unroll
      for (int i = 0; i < 8; i++) {
        int kk = half * 16 + i * 2;
        afr[mt].u[i] = ab[row * 16 + (kk >> 1)];
      }
    }
#pragma unroll
    for (int nt = 0; nt < 2; nt++) {
      int nl = wn * 32 + nt * 16 + ml;
#pragma unroll
      for (int jj = 0; jj < 8; jj++)
        bfr[nt].u[jj] = bb[nl * 16 + half * 8 + jj];
    }
#pragma unroll
    for (int mt = 0; mt < 4; mt++)
#pragma unroll
      for (int nt = 0; nt < 2; nt++)
        acc[mt][nt] = __builtin_amdgcn_wmma_f32_16x16x32_bf16(
            false, afr[mt].v, false, bfr[nt].v, (short)0, acc[mt][nt],
            false, false);
    __syncthreads();   // everyone done reading buf before it is re-staged
  }

  // epilogue: bias, scale, bf16 store
#pragma unroll
  for (int nt = 0; nt < 2; nt++) {
    int cg = col0 + wn * 32 + nt * 16 + ml;
    float bias = bp[cg];
    int h = cg >> 6, d = cg & 63;
#pragma unroll
    for (int mt = 0; mt < 4; mt++) {
#pragma unroll
      for (int r = 0; r < 8; r++) {
        int rg = row0 + wm * 64 + mt * 16 + r + half * 8;
        int bb2 = rg >> 12, ss = rg & (S_LEN - 1);
        float val = (acc[mt][nt][r] + bias) * scale;
        size_t idx;
        if (blockIdx.z == 2)   // V transposed: (B,H,D,S)
          idx = (((size_t)bb2 * H_SZ + h) * D_SZ + d) * S_LEN + ss;
        else                   // Q/K: (B,H,S,D)
          idx = (((size_t)bb2 * H_SZ + h) * S_LEN + ss) * D_SZ + d;
        outp[idx] = f2bf(val);
      }
    }
  }
}

// ---------------------------------------------------------------------------
// Kernel 2: banded local attention, flash-style online softmax.
// One block per (b,h,chunk); 8 waves, each owns 32 query rows.
// Key-block loop clipped to the in-sequence range (blocks are always fully
// valid since 32 | W), K/V async double-buffered into LDS.
// ---------------------------------------------------------------------------
__global__ __launch_bounds__(256) void local_attn_kernel(
    const uint16_t* __restrict__ qb, const uint16_t* __restrict__ kb,
    const uint16_t* __restrict__ vtw, float* __restrict__ out)
{
  __shared__ __align__(16) uint32_t k_lds[2][32 * 32];  // [buf][key][dpair]
  __shared__ __align__(16) uint32_t v_lds[2][64 * 16];  // [buf][dim][kpair]
  __shared__ uint32_t p_lds32[8 * 32 * 16];             // per-wave 32x32 bf16 P

  const int blk   = blockIdx.x;           // B*H*NC = 384
  const int chunk = blk & (NC - 1);
  const int h     = (blk >> 4) % H_SZ;
  const int b     = blk / (NC * H_SZ);

  const int tid  = threadIdx.x;
  const int wave = tid >> 5;
  const int lane = tid & 31;
  const int half = lane >> 4;
  const int nl   = lane & 15;

  const size_t bh_base = ((size_t)b * H_SZ + h) * S_LEN;
  const uint16_t* kbase  = kb + bh_base * D_SZ;
  const uint16_t* vtbase = vtw + bh_base * D_SZ;   // (B,H,D,S): same elem count
  const int qrow0 = chunk * WHALF + wave * 32;
  const int kwin0 = chunk * WHALF - WHALF;

  // Q fragments: 2 m-tiles x 2 k-steps over D=64
  Frag qf[2][2];
  const uint32_t* qw = (const uint32_t*)qb;
#pragma unroll
  for (int mt = 0; mt < 2; mt++) {
    size_t rbase = (bh_base + qrow0 + mt * 16 + nl) * 32;
#pragma unroll
    for (int ks = 0; ks < 2; ks++)
#pragma unroll
      for (int i = 0; i < 8; i++) {
        int dd = ks * 32 + half * 16 + i * 2;
        qf[mt][ks].u[i] = qw[rbase + (dd >> 1)];
      }
  }

  v8f o[2][4];
#pragma unroll
  for (int i = 0; i < 2; i++)
#pragma unroll
    for (int j = 0; j < 4; j++) o[i][j] = vzero8();
  float rmax[2][8], rsum[2][8];
#pragma unroll
  for (int mt = 0; mt < 2; mt++)
#pragma unroll
    for (int r = 0; r < 8; r++) { rmax[mt][r] = -1e30f; rsum[mt][r] = 0.0f; }

  const uint32_t kL = lds_addr_of(&k_lds[0][0]) + tid * 16;
  const uint32_t vL = lds_addr_of(&v_lds[0][0]) + tid * 16;
  uint16_t* pw = (uint16_t*)&p_lds32[wave * 32 * 16];

  // clip to in-sequence key blocks (always whole 32-key blocks)
  const int lo = (kwin0 < 0) ? -kwin0 : 0;
  const int hi = (kwin0 + WIN > S_LEN) ? (S_LEN - kwin0) : WIN;
  const int nblk = (hi - lo) >> 5;

  // issue K/V block staging for key block starting at kb0 into buffer buf
  auto issueKV = [&](int kb0, int buf) {
    int kfirst = kwin0 + kb0;
    // K tile [key][dpair]: 32 keys x 128B rows
    uint32_t goffK = (uint32_t)(kfirst + (tid >> 3)) * 128 + (tid & 7) * 16;
    async_ld16(kL + buf * 4096, goffK, kbase);
    // V tile [dim][kpair]: 64 dims, 64B (32 keys) per dim, row stride S*2
    uint32_t goffV = (uint32_t)(tid >> 2) * (S_LEN * 2) +
                     (uint32_t)(kfirst + (tid & 3) * 8) * 2;
    async_ld16(vL + buf * 4096, goffV, vtbase);
  };

  issueKV(lo, 0);
  for (int jj = 0; jj < nblk; jj++) {
    const int kb0 = lo + jj * 32;
    const int kfirst = kwin0 + kb0;
    const int buf = jj & 1;
    if (jj + 1 < nblk) { issueKV(kb0 + 32, buf ^ 1); WAIT_ASYNC(2); }
    else               { WAIT_ASYNC(0); }
    __syncthreads();

    const uint32_t* kt = k_lds[buf];
    const uint32_t* vt = v_lds[buf];

    // scores S = Q * K^T
    Frag kfr[2][2];
#pragma unroll
    for (int nt = 0; nt < 2; nt++)
#pragma unroll
      for (int ks = 0; ks < 2; ks++)
#pragma unroll
        for (int j = 0; j < 8; j++)
          kfr[nt][ks].u[j] = kt[(nt * 16 + nl) * 32 + ks * 16 + half * 8 + j];

    v8f sc[2][2];
#pragma unroll
    for (int mt = 0; mt < 2; mt++)
#pragma unroll
      for (int nt = 0; nt < 2; nt++) {
        v8f c = vzero8();
        c = __builtin_amdgcn_wmma_f32_16x16x32_bf16(
            false, qf[mt][0].v, false, kfr[nt][0].v, (short)0, c, false, false);
        c = __builtin_amdgcn_wmma_f32_16x16x32_bf16(
            false, qf[mt][1].v, false, kfr[nt][1].v, (short)0, c, false, false);
        sc[mt][nt] = c;
      }

    // band mask + online softmax
#pragma unroll
    for (int mt = 0; mt < 2; mt++) {
      float bmax[8];
#pragma unroll
      for (int r = 0; r < 8; r++) {
        int qa = qrow0 + mt * 16 + r + half * 8;
        float svals[2];
#pragma unroll
        for (int nt = 0; nt < 2; nt++) {
          int diff = (kfirst + nt * 16 + nl) - qa;
          bool band = (diff <= WHALF) && (diff >= -WHALF);
          float s = band ? sc[mt][nt][r] : -1e30f;
          sc[mt][nt][r] = s;
          svals[nt] = s;
        }
        float m2 = fmaxf(svals[0], svals[1]);
#pragma unroll
        for (int off = 8; off >= 1; off >>= 1)
          m2 = fmaxf(m2, __shfl_xor(m2, off, 32));
        bmax[r] = m2;
      }
#pragma unroll
      for (int r = 0; r < 8; r++) {
        float nm = fmaxf(rmax[mt][r], bmax[r]);
        float cf = __expf(rmax[mt][r] - nm);
        rmax[mt][r] = nm;
        float s0 = sc[mt][0][r], s1 = sc[mt][1][r];
        float p0 = (s0 > -1e29f) ? __expf(s0 - nm) : 0.0f;
        float p1 = (s1 > -1e29f) ? __expf(s1 - nm) : 0.0f;
        float ps = p0 + p1;
#pragma unroll
        for (int off = 8; off >= 1; off >>= 1)
          ps += __shfl_xor(ps, off, 32);
        rsum[mt][r] = rsum[mt][r] * cf + ps;
#pragma unroll
        for (int nt2 = 0; nt2 < 4; nt2++) o[mt][nt2][r] *= cf;
        int m = mt * 16 + r + half * 8;
        pw[m * 32 + nl]      = f2bf(p0);
        pw[m * 32 + 16 + nl] = f2bf(p1);
      }
    }

    // O += P * V
    Frag pf[2], vf[4];
#pragma unroll
    for (int mt = 0; mt < 2; mt++)
#pragma unroll
      for (int i = 0; i < 8; i++) {
        int kk = half * 16 + i * 2;
        pf[mt].u[i] = p_lds32[(wave * 32 + mt * 16 + nl) * 16 + (kk >> 1)];
      }
#pragma unroll
    for (int nt2 = 0; nt2 < 4; nt2++)
#pragma unroll
      for (int j = 0; j < 8; j++)
        vf[nt2].u[j] = vt[(nt2 * 16 + nl) * 16 + half * 8 + j];
#pragma unroll
    for (int mt = 0; mt < 2; mt++)
#pragma unroll
      for (int nt2 = 0; nt2 < 4; nt2++)
        o[mt][nt2] = __builtin_amdgcn_wmma_f32_16x16x32_bf16(
            false, pf[mt].v, false, vf[nt2].v, (short)0, o[mt][nt2],
            false, false);
    __syncthreads();
  }

  // epilogue: normalize and store f32 to (S,B,E)
#pragma unroll
  for (int mt = 0; mt < 2; mt++)
#pragma unroll
    for (int nt2 = 0; nt2 < 4; nt2++)
#pragma unroll
      for (int r = 0; r < 8; r++) {
        int srow = qrow0 + mt * 16 + r + half * 8;
        int e = h * 64 + nt2 * 16 + nl;
        out[(size_t)srow * (B_SZ * E_SZ) + (size_t)b * E_SZ + e] =
            o[mt][nt2][r] / rsum[mt][r];
      }
}

// ---------------------------------------------------------------------------
extern "C" void kernel_launch(void* const* d_in, const int* in_sizes, int n_in,
                              void* d_out, int out_size, void* d_ws,
                              size_t ws_size, hipStream_t stream) {
  const float* query = (const float*)d_in[0];
  const float* Wq = (const float*)d_in[1];
  const float* bq = (const float*)d_in[2];
  const float* Wk = (const float*)d_in[3];
  const float* bk = (const float*)d_in[4];
  const float* Wv = (const float*)d_in[5];
  const float* bv = (const float*)d_in[6];
  float* out = (float*)d_out;

  const size_t NQ = (size_t)B_SZ * H_SZ * S_LEN * D_SZ;   // 6,291,456
  uint16_t* qb  = (uint16_t*)d_ws;
  uint16_t* kbw = qb + NQ;
  uint16_t* vtw = kbw + NQ;                               // V transposed (B,H,D,S)
  uint16_t* xb  = vtw + NQ;                               // 8192 x 768 bf16
  uint16_t* wtb = xb + (size_t)(S_LEN * B_SZ) * E_SZ;     // 3 x 768 x 768 bf16

  cvt_x_kernel<<<dim3((S_LEN * B_SZ * E_SZ / 2) / 256), 256, 0, stream>>>(
      query, (uint32_t*)xb);
  cvt_w_kernel<<<dim3((E_SZ * E_SZ / 2) / 256, 3), 256, 0, stream>>>(
      Wq, Wk, Wv, (uint32_t*)wtb);

  dim3 g1((S_LEN * B_SZ) / 128, E_SZ / 128, 3);
  qkv_gemm_kernel<<<g1, 256, 0, stream>>>(xb, wtb, bq, bk, bv, qb, kbw, vtw);

  dim3 g2(B_SZ * H_SZ * NC);
  local_attn_kernel<<<g2, 256, 0, stream>>>(qb, kbw, vtw, out);
}
